// PointsEncoder_label_51367808860748
// MI455X (gfx1250) — compile-verified, matrix operand order checked by reading, not run on hardware
//
#include <hip/hip_runtime.h>
#include <hip/hip_bf16.h>
#include <stdint.h>

// ---------------------------------------------------------------------------
// PointsEncoder (PointNet++ + point-transformer) for MI455X / gfx1250.
// All dense / 1x1-conv layers run as f16 WMMA GEMMs (v_wmma_f32_16x16x32_f16,
// f32 accumulate), BN folded into weights on device. Irregular ops (FPS,
// ball query, kNN, gathers, neighbor softmax) are wave32 VALU kernels.
// ---------------------------------------------------------------------------

typedef __attribute__((ext_vector_type(16))) _Float16 v16h;
typedef __attribute__((ext_vector_type(8)))  float    v8f;

constexpr int kB  = 16;     // batch
constexpr int kN  = 8192;   // input points
constexpr int kS1 = 512;    // SA1 samples
constexpr int kS2 = 128;    // SA2 samples
constexpr int kK  = 32;     // ball-query group size
constexpr int kNK = 16;     // kNN size in transformer

// ---------------------------------------------------------------------------
// WMMA NT GEMM:  Y[M,N] = X[M,K] * W[N,K]^T + bias  (optional ReLU)
// X,W f16 row-major (K contiguous, K % 32 == 0), Y f16, bias f32.
// Block = 256 thr = 8 waves arranged 4(M) x 2(N); wave tile 32x32 -> block
// tile 128x64. M must be a multiple of 128 (true for every call site).
// ---------------------------------------------------------------------------
union FragU { v16h h; uint4 u[2]; };

__global__ __launch_bounds__(256) void gemm_nt_kernel(
    const _Float16* __restrict__ X, const _Float16* __restrict__ W,
    const float* __restrict__ bias, _Float16* __restrict__ Y,
    int M, int N, int K, int relu)
{
  (void)M;
  const int wave = threadIdx.x >> 5;
  const int lane = threadIdx.x & 31;
  const int wm = wave >> 1, wn = wave & 1;
  const int m0 = blockIdx.y * 128 + wm * 32;
  const int n0 = blockIdx.x * 64 + wn * 32;
  const int row = lane & 15;
  const int khalf = (lane & 16) ? 8 : 0;   // per CDNA5 16-bit A/B fragment layout

  const _Float16* a0 = X + (size_t)(m0 + row) * K + khalf;
  const _Float16* a1 = a0 + (size_t)16 * K;
  int nr0 = n0 + row;      if (nr0 > N - 1) nr0 = N - 1;
  int nr1 = n0 + 16 + row; if (nr1 > N - 1) nr1 = N - 1;
  const _Float16* b0 = W + (size_t)nr0 * K + khalf;
  const _Float16* b1 = W + (size_t)nr1 * K + khalf;

  v8f acc00 = {}, acc01 = {}, acc10 = {}, acc11 = {};
  for (int k = 0; k < K; k += 32) {
    FragU A0, A1, B0, B1;
    A0.u[0] = *(const uint4*)(a0 + k); A0.u[1] = *(const uint4*)(a0 + k + 16);
    A1.u[0] = *(const uint4*)(a1 + k); A1.u[1] = *(const uint4*)(a1 + k + 16);
    B0.u[0] = *(const uint4*)(b0 + k); B0.u[1] = *(const uint4*)(b0 + k + 16);
    B1.u[0] = *(const uint4*)(b1 + k); B1.u[1] = *(const uint4*)(b1 + k + 16);
    if (k + 32 < K) {   // gfx1250 global_prefetch_b8
      __builtin_prefetch((const void*)(a0 + k + 32), 0, 1);
      __builtin_prefetch((const void*)(b0 + k + 32), 0, 1);
    }
    acc00 = __builtin_amdgcn_wmma_f32_16x16x32_f16(false, A0.h, false, B0.h, (short)0, acc00, false, false);
    acc01 = __builtin_amdgcn_wmma_f32_16x16x32_f16(false, A0.h, false, B1.h, (short)0, acc01, false, false);
    acc10 = __builtin_amdgcn_wmma_f32_16x16x32_f16(false, A1.h, false, B0.h, (short)0, acc10, false, false);
    acc11 = __builtin_amdgcn_wmma_f32_16x16x32_f16(false, A1.h, false, B1.h, (short)0, acc11, false, false);
  }

  const int cn = lane & 15;
  const int cmoff = (lane & 16) ? 8 : 0;   // C/D layout: lanes 16-31 hold M=r+8
  v8f accs[2][2] = { { acc00, acc01 }, { acc10, acc11 } };
  for (int ni = 0; ni < 2; ++ni) {
    int n = n0 + ni * 16 + cn;
    if (n >= N) continue;
    float bb = bias[n];
    for (int mi = 0; mi < 2; ++mi) {
      for (int r = 0; r < 8; ++r) {
        float vv = accs[mi][ni][r] + bb;
        if (relu) vv = fmaxf(vv, 0.0f);
        Y[(size_t)(m0 + mi * 16 + cmoff + r) * N + n] = (_Float16)vv;
      }
    }
  }
}

// Fold BN (scale,bias may be null) into f16 weights, zero-pad Cin -> Cpad.
__global__ void fold_w_kernel(const float* __restrict__ W, const float* __restrict__ b,
                              const float* __restrict__ scale, const float* __restrict__ bias,
                              _Float16* __restrict__ Wf, float* __restrict__ bf,
                              int O, int Cin, int Cpad)
{
  int o = blockIdx.x * blockDim.x + threadIdx.x;
  if (o >= O) return;
  float s = scale ? scale[o] : 1.0f;
  const float* wr = W + (size_t)o * Cin;
  _Float16* wo = Wf + (size_t)o * Cpad;
  for (int c = 0; c < Cpad; ++c)
    wo[c] = (c < Cin) ? (_Float16)(wr[c] * s) : (_Float16)0.f;
  bf[o] = b[o] * s + (bias ? bias[o] : 0.f);
}

// ---------------------------------------------------------------------------
// Farthest point sampling: one block per batch, 512 threads, dist in LDS.
// Matches jax reference: start at index 0, record carry before update,
// argmax prefers lowest index on ties.
// ---------------------------------------------------------------------------
__global__ __launch_bounds__(512) void fps_kernel(const float* __restrict__ xyz,
                                                  int* __restrict__ out, int Np, int S)
{
  __shared__ float sdist[8192];
  __shared__ float rv[512];
  __shared__ int   ri[512];
  __shared__ int   sfar;
  const int b = blockIdx.x;
  const int t = threadIdx.x;
  const float* base = xyz + (size_t)b * Np * 3;
  for (int i = t; i < Np; i += 512) sdist[i] = 1e10f;
  if (t == 0) sfar = 0;
  __syncthreads();
  for (int j = 0; j < S; ++j) {
    int far = sfar;
    if (t == 0) out[b * S + j] = far;
    float cx = base[far * 3 + 0], cy = base[far * 3 + 1], cz = base[far * 3 + 2];
    float bestv = -1.f; int besti = 0x7fffffff;
    for (int i = t; i < Np; i += 512) {
      float dx = base[i * 3 + 0] - cx, dy = base[i * 3 + 1] - cy, dz = base[i * 3 + 2] - cz;
      float d = fminf(sdist[i], dx * dx + dy * dy + dz * dz);
      sdist[i] = d;
      if (d > bestv) { bestv = d; besti = i; }
    }
    rv[t] = bestv; ri[t] = besti;
    __syncthreads();
    for (int s = 256; s > 0; s >>= 1) {
      if (t < s) {
        if (rv[t + s] > rv[t] || (rv[t + s] == rv[t] && ri[t + s] < ri[t])) {
          rv[t] = rv[t + s]; ri[t] = ri[t + s];
        }
      }
      __syncthreads();
    }
    if (t == 0) sfar = ri[0];
    __syncthreads();
  }
}

__global__ void gather_xyz_kernel(const float* __restrict__ xyz, const int* __restrict__ fidx,
                                  float* __restrict__ out, int Np, int S, int total)
{
  int g = blockIdx.x * blockDim.x + threadIdx.x;
  if (g >= total) return;
  int b = g / S;
  int j = fidx[g];
  const float* p = xyz + ((size_t)b * Np + j) * 3;
  out[(size_t)g * 3 + 0] = p[0];
  out[(size_t)g * 3 + 1] = p[1];
  out[(size_t)g * 3 + 2] = p[2];
}

// CUDA-semantics ball query: first 32 in-ball points in index order, pad with
// first found (or index 0 when none).
__global__ void ball_query_kernel(const float* __restrict__ xyz, const float* __restrict__ nxyz,
                                  int* __restrict__ idx, int Np, int S, float r2, int total)
{
  int g = blockIdx.x * blockDim.x + threadIdx.x;
  if (g >= total) return;
  int b = g / S;
  const float* q = nxyz + (size_t)g * 3;
  const float* base = xyz + (size_t)b * Np * 3;
  int* o = idx + (size_t)g * kK;
  float qx = q[0], qy = q[1], qz = q[2];
  int cnt = 0;
  for (int i = 0; i < Np && cnt < kK; ++i) {
    float dx = base[i * 3 + 0] - qx, dy = base[i * 3 + 1] - qy, dz = base[i * 3 + 2] - qz;
    if (dx * dx + dy * dy + dz * dz < r2) o[cnt++] = i;
  }
  int first = (cnt > 0) ? o[0] : 0;
  for (int i = cnt; i < kK; ++i) o[i] = first;
}

// Build grouped features: [recentered xyz(3) | gathered point feats(Cp) | 0-pad] -> f16 [total, Cpad]
template <typename T>
__global__ void group_kernel(const float* __restrict__ xyz, const float* __restrict__ nxyz,
                             const int* __restrict__ idx, const T* __restrict__ pts,
                             int ptsStride, int Cp, int Cpad, _Float16* __restrict__ X,
                             int Np, int S, int total)
{
  int g = blockIdx.x * blockDim.x + threadIdx.x;
  if (g >= total) return;
  int s = (g >> 5) % S;
  int b = g / (S * kK);
  int j = idx[g];
  const float* cen = nxyz + ((size_t)b * S + s) * 3;
  const float* pj  = xyz + ((size_t)b * Np + j) * 3;
  _Float16* o = X + (size_t)g * Cpad;
  o[0] = (_Float16)(pj[0] - cen[0]);
  o[1] = (_Float16)(pj[1] - cen[1]);
  o[2] = (_Float16)(pj[2] - cen[2]);
  const T* f = pts + ((size_t)b * Np + j) * ptsStride;
  for (int c = 0; c < Cp; ++c) o[3 + c] = (_Float16)(float)f[c];
  for (int c = 3 + Cp; c < Cpad; ++c) o[c] = (_Float16)0.f;
}

__global__ void maxpool32_kernel(const _Float16* __restrict__ X, _Float16* __restrict__ Y,
                                 int C, int total)
{
  int g = blockIdx.x * blockDim.x + threadIdx.x;
  if (g >= total) return;
  int c = g % C, r = g / C;
  const _Float16* p = X + ((size_t)r * kK) * C + c;
  float m = -3e38f;
  for (int k = 0; k < kK; ++k) m = fmaxf(m, (float)p[(size_t)k * C]);
  Y[g] = (_Float16)m;
}

__global__ void concat_label_kernel(const _Float16* __restrict__ pool, const float* __restrict__ label,
                                    _Float16* __restrict__ out, int total)
{
  int g = blockIdx.x * blockDim.x + threadIdx.x;
  if (g >= total) return;
  int b = g >> 9;   // / 512
  _Float16* o = out + (size_t)g * 160;
  const _Float16* p = pool + (size_t)g * 128;
  for (int c = 0; c < 128; ++c) o[c] = p[c];
  for (int c = 0; c < 16; ++c) o[128 + c] = (_Float16)label[b * 16 + c];
  for (int c = 144; c < 160; ++c) o[c] = (_Float16)0.f;
}

// kNN (16 nearest incl. self), insertion sort, stable on ties (matches top_k order).
__global__ void knn_kernel(const float* __restrict__ pos, int* __restrict__ idx, int n, int total)
{
  int g = blockIdx.x * blockDim.x + threadIdx.x;
  if (g >= total) return;
  int b = g / n, i = g % n;
  const float* base = pos + (size_t)b * n * 3;
  float qx = base[i * 3 + 0], qy = base[i * 3 + 1], qz = base[i * 3 + 2];
  float hd[kNK]; int hix[kNK]; int cnt = 0;
  for (int j = 0; j < n; ++j) {
    float dx = base[j * 3 + 0] - qx, dy = base[j * 3 + 1] - qy, dz = base[j * 3 + 2] - qz;
    float d2 = dx * dx + dy * dy + dz * dz;
    if (cnt < kNK) {
      int p = cnt;
      while (p > 0 && hd[p - 1] > d2) { hd[p] = hd[p - 1]; hix[p] = hix[p - 1]; --p; }
      hd[p] = d2; hix[p] = j; ++cnt;
    } else if (d2 < hd[kNK - 1]) {
      int p = kNK - 1;
      while (p > 0 && hd[p - 1] > d2) { hd[p] = hd[p - 1]; hix[p] = hix[p - 1]; --p; }
      hd[p] = d2; hix[p] = j;
    }
  }
  int* o = idx + (size_t)g * kNK;
  for (int k = 0; k < kNK; ++k) o[k] = hix[k];
}

// pos_rel = pos[i] - pos[idx[i,k]] padded to 32ch f16
__global__ void posrel_kernel(const float* __restrict__ pos, const int* __restrict__ knn,
                              _Float16* __restrict__ X, int n, int total)
{
  int g = blockIdx.x * blockDim.x + threadIdx.x;
  if (g >= total) return;
  int k = g & 15, r = g >> 4;
  int b = r / n;
  int j = knn[(size_t)r * kNK + k];
  const float* pi = pos + (size_t)r * 3;
  const float* pj = pos + ((size_t)b * n + j) * 3;
  _Float16* o = X + (size_t)g * 32;
  o[0] = (_Float16)(pi[0] - pj[0]);
  o[1] = (_Float16)(pi[1] - pj[1]);
  o[2] = (_Float16)(pi[2] - pj[2]);
  for (int c = 3; c < 32; ++c) o[c] = (_Float16)0.f;
}

// attn_in = q[i] - k[idx[i,k]] + pos_emb
__global__ void attnin_kernel(const _Float16* __restrict__ q, const _Float16* __restrict__ kf,
                              const _Float16* __restrict__ pe, const int* __restrict__ knn,
                              _Float16* __restrict__ X, int n, int total)
{
  int g = blockIdx.x * blockDim.x + threadIdx.x;
  if (g >= total) return;
  int k = g & 15, r = g >> 4;
  int b = r / n;
  int j = knn[(size_t)r * kNK + k];
  int jr = b * n + j;
  const _Float16* qq = q + (size_t)r * 64;
  const _Float16* kk = kf + (size_t)jr * 64;
  const _Float16* pp = pe + (size_t)g * 64;
  _Float16* o = X + (size_t)g * 64;
  for (int c = 0; c < 64; ++c)
    o[c] = (_Float16)((float)qq[c] - (float)kk[c] + (float)pp[c]);
}

// softmax over the 16 neighbors (per channel) fused with agg = sum attn*(v+pos_emb)
__global__ void attn_agg_kernel(const _Float16* __restrict__ logit, const _Float16* __restrict__ pe,
                                const _Float16* __restrict__ v, _Float16* __restrict__ agg, int total)
{
  int g = blockIdx.x * blockDim.x + threadIdx.x;
  if (g >= total) return;
  int c = g & 63, r = g >> 6;
  const _Float16* L = logit + ((size_t)r * kNK) * 64 + c;
  const _Float16* P = pe    + ((size_t)r * kNK) * 64 + c;
  float lv[kNK];
  float mx = -3e38f;
  for (int k = 0; k < kNK; ++k) { lv[k] = (float)L[(size_t)k * 64]; mx = fmaxf(mx, lv[k]); }
  float s = 0.f;
  for (int k = 0; k < kNK; ++k) { lv[k] = __expf(lv[k] - mx); s += lv[k]; }
  float inv = 1.f / s;
  float vv = (float)v[g];
  float a = 0.f;
  for (int k = 0; k < kNK; ++k) a += lv[k] * (vv + (float)P[(size_t)k * 64]);
  agg[g] = (_Float16)(a * inv);
}

// out[r, c<cin] = delta[r,c] + ident[r,c]; pad zeros (identity stride == cinPad)
__global__ void resadd_kernel(const _Float16* __restrict__ delta, const _Float16* __restrict__ ident,
                              _Float16* __restrict__ out, int cin, int cinPad, int total)
{
  int g = blockIdx.x * blockDim.x + threadIdx.x;
  if (g >= total) return;
  int c = g % cinPad, r = g / cinPad;
  _Float16 res;
  if (c < cin)
    res = (_Float16)((float)delta[(size_t)r * cin + c] + (float)ident[(size_t)r * cinPad + c]);
  else
    res = (_Float16)0.f;
  out[(size_t)r * cinPad + c] = res;
}

// SA3 group_all input: [xyz(3) | l2(256) | 0-pad] -> [2048, 288]
__global__ void concat_sa3_kernel(const float* __restrict__ xyz, const _Float16* __restrict__ feat,
                                  _Float16* __restrict__ X, int total)
{
  int g = blockIdx.x * blockDim.x + threadIdx.x;
  if (g >= total) return;
  _Float16* o = X + (size_t)g * 288;
  const float* p = xyz + (size_t)g * 3;
  o[0] = (_Float16)p[0]; o[1] = (_Float16)p[1]; o[2] = (_Float16)p[2];
  const _Float16* f = feat + (size_t)g * 256;
  for (int c = 0; c < 256; ++c) o[3 + c] = f[c];
  for (int c = 259; c < 288; ++c) o[c] = (_Float16)0.f;
}

// final max over the 128 points per batch -> (B,512,1) f32 output
__global__ void maxpool_final_kernel(const _Float16* __restrict__ X, float* __restrict__ out, int total)
{
  int g = blockIdx.x * blockDim.x + threadIdx.x;
  if (g >= total) return;
  int c = g & 511, b = g >> 9;
  const _Float16* p = X + ((size_t)b * 128) * 512 + c;
  float m = -3e38f;
  for (int k = 0; k < 128; ++k) m = fmaxf(m, (float)p[(size_t)k * 512]);
  out[g] = m;
}

// ---------------------------------------------------------------------------
// Host-side parameter walking (handles both jax sorted-key pytree flattening
// and plain dict-insertion-order flattening, detected via in_sizes).
// ---------------------------------------------------------------------------
struct DenseP { const float *W, *b; };
struct LayerP { const float *W, *b, *scale, *bias; };
struct TransP { DenseP ls, cq, ck, cv, pos1, attn1, le; LayerP pos0, attn0; };
struct Cursor { void* const* din; int i;
  const float* next() { return (const float*)din[i++]; } };

static void read_dense(Cursor& c, DenseP& d) { d.W = c.next(); d.b = c.next(); }
static void read_layer_sorted(Cursor& c, LayerP& l) { l.W = c.next(); l.b = c.next(); l.bias = c.next(); l.scale = c.next(); }
static void read_layer_ins(Cursor& c, LayerP& l)    { l.W = c.next(); l.b = c.next(); l.scale = c.next(); l.bias = c.next(); }
static void read_trans_sorted(Cursor& c, TransP& t) {
  read_layer_sorted(c, t.attn0); read_dense(c, t.attn1);
  read_dense(c, t.ck); read_dense(c, t.cq); read_dense(c, t.cv);
  read_dense(c, t.le); read_dense(c, t.ls);
  read_layer_sorted(c, t.pos0); read_dense(c, t.pos1);
}
static void read_trans_ins(Cursor& c, TransP& t) {
  read_dense(c, t.ls); read_dense(c, t.cq); read_dense(c, t.ck); read_dense(c, t.cv);
  read_layer_ins(c, t.pos0); read_dense(c, t.pos1);
  read_layer_ins(c, t.attn0); read_dense(c, t.attn1); read_dense(c, t.le);
}

extern "C" void kernel_launch(void* const* d_in, const int* in_sizes, int n_in,
                              void* d_out, int out_size, void* d_ws, size_t ws_size,
                              hipStream_t stream)
{
  (void)out_size; (void)ws_size;
  // ---- locate inputs ----
  const float *pc, *label; int pbase;
  if (in_sizes[0] == kB * 16) {        // sorted top-level: label, params..., point_cloud
    label = (const float*)d_in[0]; pc = (const float*)d_in[n_in - 1]; pbase = 1;
  } else {                             // insertion order: point_cloud, label, params...
    pc = (const float*)d_in[0]; label = (const float*)d_in[1]; pbase = 2;
  }
  bool sorted_mode = (in_sizes[pbase + 12] == 128 * 147); // sa2.l0.W vs t1.ls.W (64*144)
  Cursor cur{d_in, pbase};
  LayerP sa1[3], sa2[3], sa3[3]; TransP t1, t2;
  if (sorted_mode) {
    for (int i = 0; i < 3; ++i) read_layer_sorted(cur, sa1[i]);
    for (int i = 0; i < 3; ++i) read_layer_sorted(cur, sa2[i]);
    for (int i = 0; i < 3; ++i) read_layer_sorted(cur, sa3[i]);
    read_trans_sorted(cur, t1); read_trans_sorted(cur, t2);
  } else {
    for (int i = 0; i < 3; ++i) read_layer_ins(cur, sa1[i]);
    read_trans_ins(cur, t1);
    for (int i = 0; i < 3; ++i) read_layer_ins(cur, sa2[i]);
    read_trans_ins(cur, t2);
    for (int i = 0; i < 3; ++i) read_layer_ins(cur, sa3[i]);
  }

  // ---- carve workspace ----
  char* ws = (char*)d_ws; size_t off = 0;
  auto carve = [&](size_t bytes) -> char* {
    char* p = ws + off; off += (bytes + 255) & ~(size_t)255; return p;
  };
  _Float16* arenaA = (_Float16*)carve(64ull << 20);   // GEMM ping
  _Float16* arenaB = (_Float16*)carve(64ull << 20);   // GEMM pong
  int*      fps1   = (int*)carve((size_t)kB * kS1 * 4);
  int*      ball1  = (int*)carve((size_t)kB * kS1 * kK * 4);
  float*    nxyz1  = (float*)carve((size_t)kB * kS1 * 3 * 4);
  _Float16* l1pool = (_Float16*)carve((size_t)kB * kS1 * 128 * 2);
  _Float16* l1buf  = (_Float16*)carve((size_t)kB * kS1 * 160 * 2);
  int*      knnbuf = (int*)carve((size_t)kB * kS1 * kNK * 4);
  _Float16* hbuf   = (_Float16*)carve((size_t)kB * kS1 * 64 * 2);
  _Float16* qbuf   = (_Float16*)carve((size_t)kB * kS1 * 64 * 2);
  _Float16* kbuf   = (_Float16*)carve((size_t)kB * kS1 * 64 * 2);
  _Float16* vbuf   = (_Float16*)carve((size_t)kB * kS1 * 64 * 2);
  _Float16* pebuf  = (_Float16*)carve((size_t)kB * kS1 * kNK * 64 * 2);
  _Float16* aggbuf = (_Float16*)carve((size_t)kB * kS1 * 64 * 2);
  int*      fps2   = (int*)carve((size_t)kB * kS2 * 4);
  int*      ball2  = (int*)carve((size_t)kB * kS2 * kK * 4);
  float*    nxyz2  = (float*)carve((size_t)kB * kS2 * 3 * 4);
  _Float16* l2pool = (_Float16*)carve((size_t)kB * kS2 * 256 * 2);
  _Float16* l2buf  = (_Float16*)carve((size_t)kB * kS2 * 256 * 2);
  _Float16* wscr   = (_Float16*)carve((size_t)512 * 512 * 2);
  float*    bscr   = (float*)carve(512 * 4);

  auto dense = [&](const float* W, const float* b, const float* scale, const float* bias,
                   const _Float16* Xp, _Float16* Yp, int M, int O, int Cin, int Cpad, int relu) {
    fold_w_kernel<<<dim3((O + 127) / 128), dim3(128), 0, stream>>>(W, b, scale, bias, wscr, bscr, O, Cin, Cpad);
    gemm_nt_kernel<<<dim3((O + 63) / 64, M / 128), dim3(256), 0, stream>>>(Xp, wscr, bscr, Yp, M, O, Cpad, relu);
  };
  auto dl = [&](const LayerP& L, const _Float16* Xp, _Float16* Yp, int M, int O, int Cin, int Cpad) {
    dense(L.W, L.b, L.scale, L.bias, Xp, Yp, M, O, Cin, Cpad, 1);
  };
  auto dd = [&](const DenseP& D, const _Float16* Xp, _Float16* Yp, int M, int O, int Cin, int Cpad) {
    dense(D.W, D.b, nullptr, nullptr, Xp, Yp, M, O, Cin, Cpad, 0);
  };

  auto transformer = [&](const TransP& T, const _Float16* xin, const float* pos,
                         int n, int cin, int cinPad, _Float16* xout) {
    int rows = kB * n, nr = rows * kNK;
    dd(T.ls, xin, hbuf, rows, 64, cin, cinPad);
    knn_kernel<<<dim3((rows + 127) / 128), dim3(128), 0, stream>>>(pos, knnbuf, n, rows);
    dd(T.cq, hbuf, qbuf, rows, 64, 64, 64);
    dd(T.ck, hbuf, kbuf, rows, 64, 64, 64);
    dd(T.cv, hbuf, vbuf, rows, 64, 64, 64);
    posrel_kernel<<<dim3((nr + 127) / 128), dim3(128), 0, stream>>>(pos, knnbuf, arenaA, n, nr);
    dense(T.pos0.W, T.pos0.b, T.pos0.scale, T.pos0.bias, arenaA, arenaB, nr, 64, 3, 32, 1);
    dd(T.pos1, arenaB, pebuf, nr, 64, 64, 64);
    attnin_kernel<<<dim3((nr + 127) / 128), dim3(128), 0, stream>>>(qbuf, kbuf, pebuf, knnbuf, arenaA, n, nr);
    dense(T.attn0.W, T.attn0.b, T.attn0.scale, T.attn0.bias, arenaA, arenaB, nr, 256, 64, 64, 1);
    dd(T.attn1, arenaB, arenaA, nr, 64, 256, 256);   // logits
    attn_agg_kernel<<<dim3((rows * 64 + 127) / 128), dim3(128), 0, stream>>>(arenaA, pebuf, vbuf, aggbuf, rows * 64);
    dd(T.le, aggbuf, arenaB, rows, cin, 64, 64);     // delta
    resadd_kernel<<<dim3((rows * cinPad + 127) / 128), dim3(128), 0, stream>>>(arenaB, xin, xout, cin, cinPad, rows * cinPad);
  };

  // ================= SA1 =================
  fps_kernel<<<dim3(kB), dim3(512), 0, stream>>>(pc, fps1, kN, kS1);
  gather_xyz_kernel<<<dim3((kB * kS1 + 127) / 128), dim3(128), 0, stream>>>(pc, fps1, nxyz1, kN, kS1, kB * kS1);
  ball_query_kernel<<<dim3((kB * kS1 + 63) / 64), dim3(64), 0, stream>>>(pc, nxyz1, ball1, kN, kS1, 0.04f, kB * kS1);
  {
    int tot = kB * kS1 * kK;  // 262144 rows
    group_kernel<float><<<dim3((tot + 127) / 128), dim3(128), 0, stream>>>(pc, nxyz1, ball1, pc, 3, 3, 32, arenaA, kN, kS1, tot);
    dl(sa1[0], arenaA, arenaB, tot, 64, 6, 32);
    dl(sa1[1], arenaB, arenaA, tot, 64, 64, 64);
    dl(sa1[2], arenaA, arenaB, tot, 128, 64, 64);
    maxpool32_kernel<<<dim3((kB * kS1 * 128 + 127) / 128), dim3(128), 0, stream>>>(arenaB, l1pool, 128, kB * kS1 * 128);
  }
  concat_label_kernel<<<dim3((kB * kS1 + 127) / 128), dim3(128), 0, stream>>>(l1pool, label, l1buf, kB * kS1);

  // ================= T1 =================
  transformer(t1, l1buf, nxyz1, kS1, 144, 160, l1buf);

  // ================= SA2 =================
  fps_kernel<<<dim3(kB), dim3(512), 0, stream>>>(nxyz1, fps2, kS1, kS2);
  gather_xyz_kernel<<<dim3((kB * kS2 + 127) / 128), dim3(128), 0, stream>>>(nxyz1, fps2, nxyz2, kS1, kS2, kB * kS2);
  ball_query_kernel<<<dim3((kB * kS2 + 63) / 64), dim3(64), 0, stream>>>(nxyz1, nxyz2, ball2, kS1, kS2, 0.16f, kB * kS2);
  {
    int tot = kB * kS2 * kK;  // 65536 rows
    group_kernel<_Float16><<<dim3((tot + 127) / 128), dim3(128), 0, stream>>>(nxyz1, nxyz2, ball2, l1buf, 160, 144, 160, arenaA, kS1, kS2, tot);
    dl(sa2[0], arenaA, arenaB, tot, 128, 147, 160);
    dl(sa2[1], arenaB, arenaA, tot, 128, 128, 128);
    dl(sa2[2], arenaA, arenaB, tot, 256, 128, 128);
    maxpool32_kernel<<<dim3((kB * kS2 * 256 + 127) / 128), dim3(128), 0, stream>>>(arenaB, l2pool, 256, kB * kS2 * 256);
  }

  // ================= T2 =================
  transformer(t2, l2pool, nxyz2, kS2, 256, 256, l2buf);

  // ================= SA3 (group_all) =================
  concat_sa3_kernel<<<dim3((kB * kS2 + 127) / 128), dim3(128), 0, stream>>>(nxyz2, l2buf, arenaA, kB * kS2);
  dl(sa3[0], arenaA, arenaB, kB * kS2, 256, 259, 288);
  dl(sa3[1], arenaB, arenaA, kB * kS2, 512, 256, 256);
  dl(sa3[2], arenaA, arenaB, kB * kS2, 512, 512, 512);
  maxpool_final_kernel<<<dim3((kB * 512 + 127) / 128), dim3(128), 0, stream>>>(arenaB, (float*)d_out, kB * 512);
}
